// ScaledDotProductAttention_8718783611541
// MI455X (gfx1250) — matrix operand order repaired and checked by reading, not compiled, not run on hardware
//
#include <hip/hip_runtime.h>

typedef __attribute__((ext_vector_type(16))) _Float16 v16h;
typedef __attribute__((ext_vector_type(8)))  _Float16 v8h;
typedef __attribute__((ext_vector_type(4)))  _Float16 v4h;
typedef __attribute__((ext_vector_type(8)))  float    v8f;

#define S_LEN   2048
#define D_HEAD  64
#define WAVES   4
#define QTILES  128      // S_LEN / 16
#define SCALE_F 0.125f   // 1/sqrt(64)

// A-operand (16x32 f16): lane m = lane&15, hi = lane>>4.
// element i<8  -> K = hi*8 + i ; element i>=8 -> K = 16 + hi*8 + (i-8)
__device__ __forceinline__ void load_a_ops(const float* __restrict__ row, int hi,
                                           v16h& a0, v16h& a1) {
#pragma unroll
    for (int j = 0; j < 8; ++j) {
        a0[j]     = (_Float16)row[hi * 8 + j];
        a0[j + 8] = (_Float16)row[16 + hi * 8 + j];
        a1[j]     = (_Float16)row[32 + hi * 8 + j];
        a1[j + 8] = (_Float16)row[48 + hi * 8 + j];
    }
}

// B-operand (32x16 f16): lane n = lane&15, hi = lane>>4.
// element i<8  -> K = hi*8 + i ; element i>=8 -> K = 16 + hi*8 + (i-8)
__device__ __forceinline__ void load_b_kT(const float* __restrict__ krow, int hi,
                                          v16h& b0, v16h& b1) {
#pragma unroll
    for (int j = 0; j < 8; ++j) {
        b0[j]     = (_Float16)krow[hi * 8 + j];
        b0[j + 8] = (_Float16)krow[16 + hi * 8 + j];
        b1[j]     = (_Float16)krow[32 + hi * 8 + j];
        b1[j + 8] = (_Float16)krow[48 + hi * 8 + j];
    }
}

__global__ void sdpa_fwd_kernel(const float* __restrict__ Q,
                                const float* __restrict__ K,
                                const float* __restrict__ V,
                                float* __restrict__ ctx,
                                float* __restrict__ prob) {
    extern __shared__ char smem_raw[];
    const int lane = threadIdx.x & 31;
    const int wave = (int)(threadIdx.x >> 5);
    _Float16* eS = (_Float16*)smem_raw + (size_t)wave * 16 * S_LEN;  // [16][2048] f16

    const int tile = (int)blockIdx.x * WAVES + wave;   // 0..8191
    const int bh   = tile >> 7;                        // head index 0..63
    const int qt   = tile & (QTILES - 1);              // query tile 0..127
    const int n16  = lane & 15;
    const int hi   = lane >> 4;

    const float* Qbase = Q + ((size_t)bh * S_LEN + (size_t)qt * 16) * D_HEAD;
    const float* Kbase = K + (size_t)bh * S_LEN * D_HEAD;
    const float* Vbase = V + (size_t)bh * S_LEN * D_HEAD;

    // Q tile A-operands (persist across both passes)
    v16h qa0, qa1;
    load_a_ops(Qbase + (size_t)n16 * D_HEAD, hi, qa0, qa1);

    // ----------------- pass 1: exact row max ------------------------------
    float rmax[8];
#pragma unroll
    for (int r = 0; r < 8; ++r) rmax[r] = -3.0e38f;

    for (int kt = 0; kt <= qt; ++kt) {
        if (kt + 1 <= qt)   // head start for next K tile (global_prefetch_b8)
            __builtin_prefetch(Kbase + (size_t)((kt + 1) * 16 + n16) * D_HEAD + hi * 16, 0, 3);
        v16h b0, b1;
        load_b_kT(Kbase + (size_t)(kt * 16 + n16) * D_HEAD, hi, b0, b1);
        v8f c = {};
        c = __builtin_amdgcn_wmma_f32_16x16x32_f16(false, qa0, false, b0, (short)0, c, false, false);
        c = __builtin_amdgcn_wmma_f32_16x16x32_f16(false, qa1, false, b1, (short)0, c, false, false);
        const int col = kt * 16 + n16;
#pragma unroll
        for (int r = 0; r < 8; ++r) {
            const int qrow = qt * 16 + hi * 8 + r;     // C layout: M = hi*8 + r
            float s = c[r] * SCALE_F;
            if (col > qrow) s = -1.0e9f;               // causal mask
            rmax[r] = fmaxf(rmax[r], s);
        }
    }
#pragma unroll
    for (int sh = 1; sh <= 8; sh <<= 1) {
#pragma unroll
        for (int r = 0; r < 8; ++r)
            rmax[r] = fmaxf(rmax[r], __shfl_xor(rmax[r], sh, 32));
    }

    // ----------------- pass 2: e = exp(s-max) -> LDS (f16), row sums -------
    float rsum[8];
#pragma unroll
    for (int r = 0; r < 8; ++r) rsum[r] = 0.0f;

    for (int kt = 0; kt <= qt; ++kt) {
        if (kt + 1 <= qt)
            __builtin_prefetch(Kbase + (size_t)((kt + 1) * 16 + n16) * D_HEAD + hi * 16, 0, 3);
        v16h b0, b1;
        load_b_kT(Kbase + (size_t)(kt * 16 + n16) * D_HEAD, hi, b0, b1);
        v8f c = {};
        c = __builtin_amdgcn_wmma_f32_16x16x32_f16(false, qa0, false, b0, (short)0, c, false, false);
        c = __builtin_amdgcn_wmma_f32_16x16x32_f16(false, qa1, false, b1, (short)0, c, false, false);
        const int col = kt * 16 + n16;
#pragma unroll
        for (int r = 0; r < 8; ++r) {
            const int qrow = qt * 16 + hi * 8 + r;
            float s = c[r] * SCALE_F;
            if (col > qrow) s = -1.0e9f;
            float e = __expf(s - rmax[r]);             // masked -> exactly 0
            rsum[r] += e;
            eS[(size_t)(hi * 8 + r) * S_LEN + col] = (_Float16)e;
        }
    }
#pragma unroll
    for (int sh = 1; sh <= 8; sh <<= 1) {
#pragma unroll
        for (int r = 0; r < 8; ++r)
            rsum[r] += __shfl_xor(rsum[r], sh, 32);
    }
    float rinv[8];
#pragma unroll
    for (int r = 0; r < 8; ++r) rinv[r] = 1.0f / rsum[r];

    // zero the 16-column stripe after the causal limit (chunk alignment for pass 3)
    if (qt < QTILES - 1) {
        const int base = (qt + 1) * 16;
#pragma unroll
        for (int i = 0; i < 8; ++i) {
            const int idx = lane + i * 32;             // 0..255
            eS[(size_t)(idx >> 4) * S_LEN + base + (idx & 15)] = (_Float16)0.0f;
        }
    }

    // make LDS writes visible before re-reading as WMMA A-operands
    asm volatile("s_wait_dscnt 0" ::: "memory");

    // ----------------- pass 3: context = (P_exp @ V) * rinv -----------------
    v8f acc[4] = {{}, {}, {}, {}};
    const int nchunks = ((qt + 1) * 16 + 31) >> 5;
    for (int ch = 0; ch < nchunks; ++ch) {
        const int kb = ch * 32;
        if (ch + 1 < nchunks)   // prefetch row 16*hi of the next V chunk
            __builtin_prefetch(Vbase + (size_t)(kb + 32 + hi * 16) * D_HEAD + n16, 0, 3);
        v16h a;
        {
            const _Float16* p = eS + (size_t)n16 * S_LEN + kb + hi * 8;
            v8h lo  = *(const v8h*)p;             // ds_load_b128: K = kb+hi*8 .. +7
            v8h hi8 = *(const v8h*)(p + 16);      // ds_load_b128: K = kb+16+hi*8 ..
#pragma unroll
            for (int j = 0; j < 8; ++j) { a[j] = lo[j]; a[j + 8] = hi8[j]; }
        }
#pragma unroll
        for (int nt = 0; nt < 4; ++nt) {
            v16h b;
            const float* vcol = Vbase + (size_t)(kb + hi * 8) * D_HEAD + nt * 16 + n16;
#pragma unroll
            for (int j = 0; j < 8; ++j) {
                b[j]     = (_Float16)vcol[(size_t)j * D_HEAD];
                b[j + 8] = (_Float16)vcol[(size_t)(16 + j) * D_HEAD];
            }
            acc[nt] = __builtin_amdgcn_wmma_f32_16x16x32_f16(false, a, false, b,
                                                             (short)0, acc[nt], false, false);
        }
    }

    // store context: C layout lane n16 -> d column, VGPR r -> row hi*8+r
    float* ctile = ctx + ((size_t)bh * S_LEN + (size_t)qt * 16) * D_HEAD;
#pragma unroll
    for (int nt = 0; nt < 4; ++nt) {
#pragma unroll
        for (int r = 0; r < 8; ++r) {
            const int m = hi * 8 + r;
            ctile[(size_t)m * D_HEAD + nt * 16 + n16] = acc[nt][r] * rinv[r];
        }
    }

    // ----------------- stream attn_prob (normalized; zeros past causal tile) -
    const int limit = (qt + 1) * 16;   // multiple of 16; float4 groups never straddle
    float* ptile = prob + ((size_t)bh * S_LEN + (size_t)qt * 16) * S_LEN;
    for (int m = 0; m < 16; ++m) {
        const float invm = __shfl(rinv[m & 7], (m >> 3) * 16, 32);
        float* prow = ptile + (size_t)m * S_LEN;
        const _Float16* erow = eS + (size_t)m * S_LEN;
        for (int c4 = lane * 4; c4 < S_LEN; c4 += 128) {
            float4 o;
            if (c4 < limit) {
                v4h e4 = *(const v4h*)(erow + c4);
                o.x = (float)e4[0] * invm;
                o.y = (float)e4[1] * invm;
                o.z = (float)e4[2] * invm;
                o.w = (float)e4[3] * invm;
            } else {
                o.x = 0.0f; o.y = 0.0f; o.z = 0.0f; o.w = 0.0f;
            }
            *(float4*)(prow + c4) = o;   // global_store_b128
        }
    }
}

extern "C" void kernel_launch(void* const* d_in, const int* in_sizes, int n_in,
                              void* d_out, int out_size, void* d_ws, size_t ws_size,
                              hipStream_t stream) {
    (void)in_sizes; (void)n_in; (void)out_size; (void)d_ws; (void)ws_size;
    const float* Q = (const float*)d_in[0];
    const float* K = (const float*)d_in[1];
    const float* V = (const float*)d_in[2];
    // d_in[3] = causal mask (bool) — mask is computed analytically in-kernel.

    float* ctx  = (float*)d_out;                                   // [4,16,2048,64]
    float* prob = (float*)d_out + (size_t)4 * 16 * 2048 * 64;      // [4,16,2048,2048]

    const size_t shmem = (size_t)WAVES * 16 * S_LEN * sizeof(_Float16);  // 256 KB
    hipFuncSetAttribute((const void*)sdpa_fwd_kernel,
                        hipFuncAttributeMaxDynamicSharedMemorySize, (int)shmem);

    // 8192 query tiles total / 4 waves per block
    dim3 grid(2048), block(WAVES * 32);
    sdpa_fwd_kernel<<<grid, block, shmem, stream>>>(Q, K, V, ctx, prob);
}
